// MMFLinearv1_7791070675567
// MI455X (gfx1250) — 8-bit matrix operand order repaired and checked by reading, compiled, not run on hardware
//
#include <hip/hip_runtime.h>
#include <cstdint>
#include <cstddef>

typedef int v8i __attribute__((ext_vector_type(8)));

typedef int v4i_b __attribute__((vector_size(16)));
typedef __attribute__((address_space(1))) v4i_b* as1_v4i_p;
typedef __attribute__((address_space(3))) v4i_b* as3_v4i_p;

#define EPS_F 1e-8f

#ifndef __has_builtin
#define __has_builtin(x) 0
#endif

#if __has_builtin(__builtin_amdgcn_global_load_async_to_lds_b128)
#define HAVE_ASYNC_LDS 1
#else
#define HAVE_ASYNC_LDS 0
#endif

__device__ __forceinline__ void cp16(void* lds_dst, const void* gsrc) {
#if HAVE_ASYNC_LDS
    __builtin_amdgcn_global_load_async_to_lds_b128(
        (as1_v4i_p)(void*)gsrc,
        (as3_v4i_p)lds_dst,
         0,  0);
#else
    *(int4*)lds_dst = *(const int4*)gsrc;
#endif
}

__device__ __forceinline__ void wait_cp() {
#if HAVE_ASYNC_LDS
#if __has_builtin(__builtin_amdgcn_s_wait_asynccnt)
    __builtin_amdgcn_s_wait_asynccnt(0);
#else
    asm volatile("s_wait_asynccnt 0x0" ::: "memory");
#endif
#endif
}

static constexpr size_t WS_SCALES   = 0;
static constexpr size_t WS_PARTIALS = 256;
static constexpr size_t WS_ROWSCALE = 8192;
static constexpr size_t WS_WQ       = 131072;
static constexpr size_t WS_AQ       = 8388608;

__global__ void absmean_partial(const float* __restrict__ w,
                                float* __restrict__ partials, int n) {
    __shared__ float red[256];
    const int tid = threadIdx.x;
    size_t idx    = (size_t)blockIdx.x * 256 + tid;
    size_t stride = (size_t)gridDim.x * 256;
    float s = 0.0f;
    for (size_t i = idx; i < (size_t)n; i += stride) s += fabsf(w[i]);
    red[tid] = s;
    __syncthreads();
    for (int off = 128; off > 0; off >>= 1) {
        if (tid < off) red[tid] += red[tid + off];
        __syncthreads();
    }
    if (tid == 0) partials[blockIdx.x] = red[0];
}

__global__ void absmean_final(const float* __restrict__ partials,
                              float* __restrict__ scales, float inv_n) {
    __shared__ float red[256];
    const int tid = threadIdx.x;
    red[tid] = partials[tid];
    __syncthreads();
    for (int off = 128; off > 0; off >>= 1) {
        if (tid < off) red[tid] += red[tid + off];
        __syncthreads();
    }
    if (tid == 0) {
        float m   = fmaxf(red[0] * inv_n, EPS_F);
        scales[0] = 1.0f / m;
        scales[1] = m;
    }
}

__global__ void quant_weight(const float4* __restrict__ w4,
                             const float* __restrict__ scales,
                             char4* __restrict__ wq4, int n4) {
    const int idx = blockIdx.x * 256 + threadIdx.x;
    if (idx >= n4) return;
    const float sw = scales[0];
    float4 v = w4[idx];
    char4 q;
    int t;
    t = (int)rintf(sw * v.x); t = t < -1 ? -1 : (t > 1 ? 1 : t); q.x = (char)t;
    t = (int)rintf(sw * v.y); t = t < -1 ? -1 : (t > 1 ? 1 : t); q.y = (char)t;
    t = (int)rintf(sw * v.z); t = t < -1 ? -1 : (t > 1 ? 1 : t); q.z = (char)t;
    t = (int)rintf(sw * v.w); t = t < -1 ? -1 : (t > 1 ? 1 : t); q.w = (char)t;
    wq4[idx] = q;
}

__global__ void __launch_bounds__(256)
norm_quant(const float* __restrict__ x, int8_t* __restrict__ aq,
           float* __restrict__ rowscale, int K) {
    const int row = blockIdx.x;
    const int tid = threadIdx.x;
    const float* xr = x + (size_t)row * K;

    float v[8];
    {
        const float4* xv = (const float4*)(xr + tid * 8);
        float4 p0 = xv[0], p1 = xv[1];
        v[0] = p0.x; v[1] = p0.y; v[2] = p0.z; v[3] = p0.w;
        v[4] = p1.x; v[5] = p1.y; v[6] = p1.z; v[7] = p1.w;
    }

    float s = 0.0f, ss = 0.0f;
#pragma unroll
    for (int i = 0; i < 8; ++i) { s += v[i]; ss += v[i] * v[i]; }

    __shared__ float r1[256];
    __shared__ float r2[256];
    r1[tid] = s; r2[tid] = ss;
    __syncthreads();
    for (int off = 128; off > 0; off >>= 1) {
        if (tid < off) { r1[tid] += r1[tid + off]; r2[tid] += r2[tid + off]; }
        __syncthreads();
    }
    const float invK = 1.0f / (float)K;
    const float mean = r1[0] * invK;
    const float var  = r2[0] * invK - mean * mean;
    const float rstd = 1.0f / sqrtf(var + EPS_F);
    __syncthreads();

    float amax = 0.0f;
#pragma unroll
    for (int i = 0; i < 8; ++i) {
        v[i] = (v[i] - mean) * rstd;
        amax = fmaxf(amax, fabsf(v[i]));
    }
    r1[tid] = amax;
    __syncthreads();
    for (int off = 128; off > 0; off >>= 1) {
        if (tid < off) r1[tid] = fmaxf(r1[tid], r1[tid + off]);
        __syncthreads();
    }
    amax = fmaxf(r1[0], EPS_F);
    const float sact = 127.0f / amax;

    int q[8];
#pragma unroll
    for (int i = 0; i < 8; ++i) {
        int t = (int)rintf(sact * v[i]);
        q[i] = t < -128 ? -128 : (t > 127 ? 127 : t);
    }
    int2 pk;
    pk.x = (q[0] & 0xff) | ((q[1] & 0xff) << 8) | ((q[2] & 0xff) << 16) | ((q[3] & 0xff) << 24);
    pk.y = (q[4] & 0xff) | ((q[5] & 0xff) << 8) | ((q[6] & 0xff) << 16) | ((q[7] & 0xff) << 24);
    *(int2*)(aq + (size_t)row * K + tid * 8) = pk;

    if (tid == 0) rowscale[row] = amax * (1.0f / 127.0f);
}

#define BM 128
#define BN 128
#define BK 128
#define BKP 144

__global__ void __launch_bounds__(256)
gemm_i8(const int8_t* __restrict__ Aq, const int8_t* __restrict__ Wq,
        const float* __restrict__ rowscale, const float* __restrict__ scales,
        const float* __restrict__ bias, float* __restrict__ out,
        int M, int N, int K) {
    __shared__ __align__(16) int8_t At[2][BM * BKP];
    __shared__ __align__(16) int8_t Bt[2][BN * BKP];

    const int tid  = threadIdx.x;
    const int wave = tid >> 5;
    const int lane = tid & 31;
    const int l16  = lane & 15;
    const int g    = lane >> 4;
    const int wm   = wave >> 1;
    const int wn   = wave & 1;

    const int n0 = blockIdx.x * BN;
    const int m0 = blockIdx.y * BM;

    v8i acc[2][4] = {};

    const int nsteps = K / BK;

    auto issue_tile = [&](int buf, int k0) {
#pragma unroll
        for (int u = 0; u < 4; ++u) {
            const int c   = u * 256 + tid;
            const int r   = c >> 3;
            const int col = (c & 7) * 16;
            cp16(&At[buf][r * BKP + col],
                 Aq + (size_t)(m0 + r) * K + k0 + col);
            cp16(&Bt[buf][r * BKP + col],
                 Wq + (size_t)(n0 + r) * K + k0 + col);
        }
    };

    issue_tile(0, 0);
    wait_cp();
    __syncthreads();

    for (int i = 0; i < nsteps; ++i) {
        const int cur = i & 1;
        if (i + 1 < nsteps) issue_tile(cur ^ 1, (i + 1) * BK);

        const int8_t* arow0 = &At[cur][(wm * 32 + l16) * BKP];
        const int8_t* arow1 = arow0 + 16 * BKP;

#pragma unroll
        for (int s = 0; s < 2; ++s) {
            v8i a[2];
#pragma unroll
            for (int ii = 0; ii < 2; ++ii) {
                const int8_t* ar = (ii ? arow1 : arow0) + s * 64;
                int2 p;
                p = *(const int2*)(ar + g * 8);      a[ii][0] = p.x; a[ii][1] = p.y;
                p = *(const int2*)(ar + 16 + g * 8); a[ii][2] = p.x; a[ii][3] = p.y;
                p = *(const int2*)(ar + 32 + g * 8); a[ii][4] = p.x; a[ii][5] = p.y;
                p = *(const int2*)(ar + 48 + g * 8); a[ii][6] = p.x; a[ii][7] = p.y;
            }
#pragma unroll
            for (int j = 0; j < 4; ++j) {
                const int8_t* brow = &Bt[cur][(wn * 64 + j * 16 + l16) * BKP] + s * 64;
                v8i b;
                int2 q;
                q = *(const int2*)(brow + g * 8);      b[0] = q.x; b[1] = q.y;
                q = *(const int2*)(brow + 16 + g * 8); b[2] = q.x; b[3] = q.y;
                q = *(const int2*)(brow + 32 + g * 8); b[4] = q.x; b[5] = q.y;
                q = *(const int2*)(brow + 48 + g * 8); b[6] = q.x; b[7] = q.y;
#pragma unroll
                for (int ii = 0; ii < 2; ++ii) {
                    acc[ii][j] = __builtin_amdgcn_wmma_i32_16x16x64_iu8(
                         true, a[ii],  true, b, acc[ii][j],
                         false,  false);
                }
            }
        }
        wait_cp();
        __syncthreads();
    }

    const float winv = scales[1];
    const int mBase = m0 + wm * 32;
    const int nBase = n0 + wn * 64;

    float bv[4];
#pragma unroll
    for (int j = 0; j < 4; ++j) bv[j] = bias[nBase + j * 16 + l16];

#pragma unroll
    for (int ii = 0; ii < 2; ++ii) {
#pragma unroll
        for (int r = 0; r < 8; ++r) {
            const int m = mBase + ii * 16 + g * 8 + r;
            const float sc = rowscale[m] * winv;
            float* orow = out + (size_t)m * N + nBase + l16;
#pragma unroll
            for (int j = 0; j < 4; ++j)
                orow[j * 16] = (float)acc[ii][j][r] * sc + bv[j];
        }
    }
}

extern "C" void kernel_launch(void* const* d_in, const int* in_sizes, int n_in,
                              void* d_out, int out_size, void* d_ws, size_t ws_size,
                              hipStream_t stream) {
    const float* x    = (const float*)d_in[0];
    const float* w    = (const float*)d_in[1];
    const float* bias = (const float*)d_in[2];
    float* out        = (float*)d_out;

    const int N = in_sizes[2];
    const int K = in_sizes[1] / N;
    const int M = in_sizes[0] / K;

    char* ws          = (char*)d_ws;
    float* scales     = (float*)(ws + WS_SCALES);
    float* partials   = (float*)(ws + WS_PARTIALS);
    float* rowscale   = (float*)(ws + WS_ROWSCALE);
    int8_t* Wq        = (int8_t*)(ws + WS_WQ);
    int8_t* Aq        = (int8_t*)(ws + WS_AQ);

    const int nw = N * K;

    absmean_partial<<<256, 256, 0, stream>>>(w, partials, nw);
    absmean_final<<<1, 256, 0, stream>>>(partials, scales, 1.0f / (float)nw);
    quant_weight<<<(nw / 4 + 255) / 256, 256, 0, stream>>>(
        (const float4*)w, scales, (char4*)Wq, nw / 4);
    norm_quant<<<M, 256, 0, stream>>>(x, Aq, rowscale, K);

    dim3 grid(N / BN, M / BM);
    gemm_i8<<<grid, 256, 0, stream>>>(Aq, Wq, rowscale, scales, bias, out, M, N, K);
}
